// RNNReadout_15891378995513
// MI455X (gfx1250) — matrix operand order repaired and checked by reading, not compiled, not run on hardware
//
#include <hip/hip_runtime.h>
#include <hip/hip_bf16.h>
#include <math.h>

// ---------------------------------------------------------------------------
// RNN readout (GRU + softmax decode), CDNA5 / gfx1250, wave32 WMMA f16->f32.
// B=512 rows -> 32 workgroups x 16 rows; each WG runs all 256 serial steps
// from LDS. Weights pre-packed to per-lane WMMA-B fragments (f16) in d_ws.
// Unrolling restrained (VGPRs < 256, no spills); activations are branch-free
// (__expf + v_rcp_f32) to keep the serial dependence chain short.
// ---------------------------------------------------------------------------

typedef __attribute__((ext_vector_type(16))) _Float16 f16x16;
typedef __attribute__((ext_vector_type(8)))  _Float16 f16x8;
typedef __attribute__((ext_vector_type(8)))  float    f32x8;

#define BATCH 512
#define TSZ   64
#define NSUB  4
#define ODIM  278
#define ODIMP 288      // padded to 32 (K) / 16 (N)
#define EDIM  256
#define CDIM  128
#define XDIM  384      // EDIM + CDIM
#define HDIM  512
#define G3    1536     // 3*HDIM

#define KC_EMB 9       // 288/32 K-chunks
#define KC_X   12      // 384/32
#define KC_H   16      // 512/32
#define NT_EMB 16      // 256/16 N-tiles
#define NT_G   96      // 1536/16
#define NT_DEC 18      // 288/16

// workspace layout (in f16 elements)
#define SZ_EMB  ((size_t)NT_EMB*KC_EMB*512)
#define SZ_GRU  ((size_t)NT_G*KC_X*512)
#define SZ_UGRU ((size_t)NT_G*KC_H*512)
#define SZ_DEC  ((size_t)NT_DEC*KC_H*512)
#define OFF_EMB  ((size_t)0)
#define OFF_GRU  (OFF_EMB + SZ_EMB)
#define OFF_UGRU (OFF_GRU + SZ_GRU)
#define OFF_DEC  (OFF_UGRU + SZ_UGRU)

// ---------------------------------------------------------------------------
// Pack f32 row-major W[K][N] -> f16 WMMA-B fragments:
//   P[((nt*Kc + kc)*32 + lane)*16 + j]
// lane 0..15  -> n = nt*16+lane,     K offsets kc*32 + {0..7, 16..23}
// lane 16..31 -> n = nt*16+lane-16,  K offsets kc*32 + {8..15, 24..31}
// Zero-pads K/N beyond the real dims.
// ---------------------------------------------------------------------------
__global__ void pack_w(const float* __restrict__ W, _Float16* __restrict__ P,
                       int K, int N, int Kc, int Nt) {
  int tid = blockIdx.x * blockDim.x + threadIdx.x;
  int total = Nt * Kc * 32;
  if (tid >= total) return;
  int lane = tid & 31;
  int kc   = (tid >> 5) % Kc;
  int nt   = tid / (32 * Kc);
  int n    = nt * 16 + (lane & 15);
  int kb   = kc * 32 + ((lane >> 4) << 3);
  _Float16* dst = P + ((size_t)tid << 4);
#pragma unroll
  for (int j = 0; j < 16; ++j) {
    int k = kb + j + ((j >> 3) << 3);
    float v = (k < K && n < N) ? W[(size_t)k * N + n] : 0.0f;
    dst[j] = (_Float16)v;
  }
}

// A fragment (16x32 f16) from LDS row-major tile [16][stride].
// lane<16:  row=lane,     K = kc*32 + {0..7, 16..23}
// lane>=16: row=lane-16,  K = kc*32 + {8..15, 24..31}
__device__ __forceinline__ f16x16 load_a_lds(const _Float16* s, int stride,
                                             int kc, int lane) {
  int m = lane & 15;
  int half = lane >> 4;
  const _Float16* p = s + m * stride + kc * 32 + half * 8;
  f16x8 lo = *(const f16x8*)p;
  f16x8 hi = *(const f16x8*)(p + 16);
  return __builtin_shufflevector(lo, hi, 0, 1, 2, 3, 4, 5, 6, 7,
                                 8, 9, 10, 11, 12, 13, 14, 15);
}

// B fragment (32x16 f16): one contiguous 32-byte load per lane.
__device__ __forceinline__ f16x16 load_b_pk(const _Float16* __restrict__ P,
                                            int nt, int kc, int Kc, int lane) {
  return *(const f16x16*)(P + ((((size_t)nt * Kc + kc) * 32 + lane) << 4));
}

#define WMMA(acc, a, b)                                                        \
  acc = __builtin_amdgcn_wmma_f32_16x16x32_f16(false, (a), false, (b),         \
                                               (short)0, (acc), false, false)

// Branch-free activations (exp + v_rcp_f32); exact saturation at +-inf.
__device__ __forceinline__ float fast_sigmoid(float x) {
  return __builtin_amdgcn_rcpf(1.0f + __expf(-x));
}
__device__ __forceinline__ float fast_tanh(float x) {
  return 1.0f - 2.0f * __builtin_amdgcn_rcpf(__expf(2.0f * x) + 1.0f);
}

// ---------------------------------------------------------------------------
// Persistent recurrence kernel: 32 blocks x 256 threads (8 wave32).
// ---------------------------------------------------------------------------
__global__ void __launch_bounds__(256, 1)
rnn_readout(const float* __restrict__ o_note, const float* __restrict__ cond,
            const float* __restrict__ b_emb, const float* __restrict__ b_gru,
            const float* __restrict__ b_dec,
            const _Float16* __restrict__ Pemb, const _Float16* __restrict__ Pgru,
            const _Float16* __restrict__ Pugru, const _Float16* __restrict__ Pdec,
            float* __restrict__ out) {
  __shared__ _Float16 s_lst[16 * ODIMP];     // softmax output (K-padded)
  __shared__ _Float16 s_x[16 * XDIM];        // [e | cond]
  __shared__ _Float16 s_h[2][16 * HDIM];     // double-buffered hidden state
  __shared__ _Float16 s_logit[16 * ODIMP];   // decoder logits / probs staging

  const int tid = threadIdx.x;
  const int lane = tid & 31;
  const int wave = tid >> 5;
  const int rowbase = blockIdx.x * 16;

  // initial lst = o_note_in[:,0,0]; zero the K-padding once (never rewritten)
#pragma unroll 1
  for (int idx = tid; idx < 16 * ODIMP; idx += 256) {
    int m = idx / ODIMP, c = idx % ODIMP;
    float v = (c < ODIM) ? o_note[(size_t)(rowbase + m) * ODIM + c] : 0.0f;
    s_lst[idx] = (_Float16)v;
  }
  __syncthreads();

  int cur = 0;
#pragma unroll 1
  for (int t = 0; t < TSZ; ++t) {
    // reference resets h to h_init (zeros) every timestep
#pragma unroll 1
    for (int idx = tid; idx < 16 * HDIM; idx += 256)
      s_h[cur][idx] = (_Float16)0.0f;
    __syncthreads();

#pragma unroll 1
    for (int i = 0; i < NSUB; ++i) {
      // ---- e = tanh(lst @ W_emb + b_emb) -> x[:,0:256] ------------------
#pragma unroll 1
      for (int nn = 0; nn < 2; ++nn) {
        int nt = wave * 2 + nn;               // 16 N-tiles over 8 waves
        f32x8 acc = {};
#pragma unroll 2
        for (int kc = 0; kc < KC_EMB; ++kc) {
          f16x16 a = load_a_lds(s_lst, ODIMP, kc, lane);
          f16x16 b = load_b_pk(Pemb, nt, kc, KC_EMB, lane);
          WMMA(acc, a, b);
        }
        int col = nt * 16 + (lane & 15);
        float bias = b_emb[col];
        int mh = (lane >> 4) * 8;
#pragma unroll
        for (int v = 0; v < 8; ++v)
          s_x[(mh + v) * XDIM + col] = (_Float16)fast_tanh(acc[v] + bias);
      }
      // cond slice -> x[:,256:384]
#pragma unroll 1
      for (int idx = tid; idx < 16 * CDIM; idx += 256) {
        int m = idx >> 7, c = idx & 127;
        float v = cond[(((size_t)(rowbase + m) * TSZ + t) * NSUB + i) * CDIM + c];
        s_x[m * XDIM + EDIM + c] = (_Float16)v;
      }
      __syncthreads();

      // ---- GRU cell: fused gx + gh, double-buffered h -------------------
      {
        int nxt = cur ^ 1;
#pragma unroll 1
        for (int q = 0; q < 4; ++q) {
          int jt = wave * 4 + q;              // hidden N-tile 0..31
          f32x8 az = {}, ar = {}, axc = {}, ahc = {};
#pragma unroll 2
          for (int kc = 0; kc < KC_X; ++kc) { // x @ W_gru (z|r|h slabs)
            f16x16 a = load_a_lds(s_x, XDIM, kc, lane);
            WMMA(az,  a, load_b_pk(Pgru, jt,      kc, KC_X, lane));
            WMMA(ar,  a, load_b_pk(Pgru, jt + 32, kc, KC_X, lane));
            WMMA(axc, a, load_b_pk(Pgru, jt + 64, kc, KC_X, lane));
          }
#pragma unroll 2
          for (int kc = 0; kc < KC_H; ++kc) { // h @ U_gru
            f16x16 a = load_a_lds(s_h[cur], HDIM, kc, lane);
            WMMA(az,  a, load_b_pk(Pugru, jt,      kc, KC_H, lane));
            WMMA(ar,  a, load_b_pk(Pugru, jt + 32, kc, KC_H, lane));
            WMMA(ahc, a, load_b_pk(Pugru, jt + 64, kc, KC_H, lane));
          }
          int col = jt * 16 + (lane & 15);
          float bz = b_gru[col], br = b_gru[HDIM + col], bc = b_gru[2 * HDIM + col];
          int mh = (lane >> 4) * 8;
#pragma unroll
          for (int v = 0; v < 8; ++v) {
            int m = mh + v;
            float hold = (float)s_h[cur][m * HDIM + col];
            float z = fast_sigmoid(az[v] + bz);
            float r = fast_sigmoid(ar[v] + br);
            float hc = fast_tanh(axc[v] + bc + r * ahc[v]);
            s_h[nxt][m * HDIM + col] = (_Float16)(z * hold + (1.0f - z) * hc);
          }
        }
        __syncthreads();
        cur = nxt;
      }

      // ---- logits = h @ W_dec + b_dec -----------------------------------
#pragma unroll 1
      for (int nt = wave; nt < NT_DEC; nt += 8) {
        f32x8 acc = {};
#pragma unroll 2
        for (int kc = 0; kc < KC_H; ++kc) {
          f16x16 a = load_a_lds(s_h[cur], HDIM, kc, lane);
          WMMA(acc, a, load_b_pk(Pdec, nt, kc, KC_H, lane));
        }
        int col = nt * 16 + (lane & 15);
        float bias = (col < ODIM) ? b_dec[col] : 0.0f;
        int mh = (lane >> 4) * 8;
#pragma unroll
        for (int v = 0; v < 8; ++v)
          s_logit[(mh + v) * ODIMP + col] = (_Float16)(acc[v] + bias);
      }
      __syncthreads();

      // ---- softmax -> lst (f16 LDS) + global output ---------------------
      {
        int row = tid >> 4, c0 = tid & 15;   // 16 threads per row
        _Float16* lr = s_logit + row * ODIMP;
        float mx = -3.0e38f;
#pragma unroll 1
        for (int c = c0; c < ODIM; c += 16) mx = fmaxf(mx, (float)lr[c]);
#pragma unroll
        for (int msk = 8; msk >= 1; msk >>= 1)
          mx = fmaxf(mx, __shfl_xor(mx, msk, 32));
        float sum = 0.0f;
#pragma unroll 1
        for (int c = c0; c < ODIM; c += 16) {
          float e = __expf((float)lr[c] - mx);
          lr[c] = (_Float16)e;               // stash exp() in place
          sum += e;
        }
#pragma unroll
        for (int msk = 8; msk >= 1; msk >>= 1) sum += __shfl_xor(sum, msk, 32);
        float inv = __builtin_amdgcn_rcpf(sum);
        float* orow = out + (((size_t)(rowbase + row) * TSZ + t) * NSUB + i) * ODIM;
#pragma unroll 1
        for (int c = c0; c < ODIM; c += 16) {
          float p = (float)lr[c] * inv;
          orow[c] = p;
          s_lst[row * ODIMP + c] = (_Float16)p;
        }
      }
      __syncthreads();
    }
  }
}

extern "C" void kernel_launch(void* const* d_in, const int* in_sizes, int n_in,
                              void* d_out, int out_size, void* d_ws, size_t ws_size,
                              hipStream_t stream) {
  (void)in_sizes; (void)n_in; (void)out_size; (void)ws_size;
  const float* o_note = (const float*)d_in[0];
  const float* cond   = (const float*)d_in[1];
  const float* W_emb  = (const float*)d_in[2];
  const float* b_emb  = (const float*)d_in[3];
  const float* W_gru  = (const float*)d_in[4];
  const float* U_gru  = (const float*)d_in[5];
  const float* b_gru  = (const float*)d_in[6];
  const float* W_dec  = (const float*)d_in[7];
  const float* b_dec  = (const float*)d_in[8];
  float* out = (float*)d_out;

  _Float16* P = (_Float16*)d_ws;   // ~3.2 MB of packed f16 fragments
  _Float16* Pemb  = P + OFF_EMB;
  _Float16* Pgru  = P + OFF_GRU;
  _Float16* Pugru = P + OFF_UGRU;
  _Float16* Pdec  = P + OFF_DEC;

  // repack weights every call (deterministic; inputs never mutated)
  pack_w<<<(NT_EMB * KC_EMB * 32 + 255) / 256, 256, 0, stream>>>(
      W_emb, Pemb, ODIM, EDIM, KC_EMB, NT_EMB);
  pack_w<<<(NT_G * KC_X * 32 + 255) / 256, 256, 0, stream>>>(
      W_gru, Pgru, XDIM, G3, KC_X, NT_G);
  pack_w<<<(NT_G * KC_H * 32 + 255) / 256, 256, 0, stream>>>(
      U_gru, Pugru, HDIM, G3, KC_H, NT_G);
  pack_w<<<(NT_DEC * KC_H * 32 + 255) / 256, 256, 0, stream>>>(
      W_dec, Pdec, HDIM, ODIM, KC_H, NT_DEC);

  rnn_readout<<<dim3(BATCH / 16), dim3(256), 0, stream>>>(
      o_note, cond, b_emb, b_gru, b_dec, Pemb, Pgru, Pugru, Pdec, out);
}
